// TPAAttention_481036337561
// MI455X (gfx1250) — matrix operand order repaired and checked by reading, not compiled, not run on hardware
//
#include <hip/hip_runtime.h>
#include <hip/hip_bf16.h>
#include <cstdint>

typedef __attribute__((ext_vector_type(16))) __bf16 v16bf;
typedef __attribute__((ext_vector_type(8)))  float  v8f;

#define SCALING_F 0.0625f      /* 256^-0.5 */
#define SOFTCAP_F 50.0f
#define NEGV (-2.3819763e38f)

static __device__ __forceinline__ unsigned short f2bf(float x) {
  union { float f; unsigned u; } v; v.f = x;
  unsigned r = v.u + 0x7FFFu + ((v.u >> 16) & 1u);   // round-to-nearest-even
  return (unsigned short)(r >> 16);
}

union Frag { v16bf v; unsigned u[8]; };

// ---- CDNA5 async copy: 16B per lane, global -> LDS, tracked by ASYNCcnt
static __device__ __forceinline__ void async_b128(unsigned lds_addr, const void* g) {
  asm volatile("global_load_async_to_lds_b128 %0, %1, off"
               :: "v"(lds_addr), "v"((unsigned long long)(size_t)g)
               : "memory");
}
static __device__ __forceinline__ void wait_async0() {
  asm volatile("s_wait_asynccnt 0" ::: "memory");
}

// ---------------------------------------------------------------- convert (plain)
__global__ __launch_bounds__(256) void cvt_f32_bf16(const float* __restrict__ s,
                                                    unsigned short* __restrict__ d, int n) {
  int i = blockIdx.x * 256 + threadIdx.x;
  int stride = gridDim.x * 256;
  for (; i < n; i += stride) d[i] = f2bf(s[i]);
}

// ---------------------------------------------------------------- convert + transpose + pad
// src: f32 [K, Nsrc] -> dst: bf16 [Npad, K]  (rows n >= Nsrc are zero)
__global__ __launch_bounds__(256) void cvt_transpose_pad(
    const float* __restrict__ s, unsigned short* __restrict__ d,
    int K, int Nsrc, int Npad)
{
  int idx = blockIdx.x * 256 + threadIdx.x;
  int total = K * Npad;
  int stride = gridDim.x * 256;
  for (; idx < total; idx += stride) {
    int n = idx / K, kk = idx - n * K;
    d[idx] = (n < Nsrc) ? f2bf(s[(size_t)kk * Nsrc + n]) : (unsigned short)0;
  }
}

// ---------------------------------------------------------------- bf16 WMMA GEMM
// C[M,N] (f32) = X[M,K] (bf16, row-major) * Wt[N,K] (bf16, pre-transposed).
// M%256==0, N%128==0, K%64==0.
// Workgroup tile 256x128 (8 waves, each 64x64 = 4x4 WMMA tiles) -> 1:1 ds_b128:wmma.
// Double-buffered async-to-LDS staging; 96 KB STATIC LDS (AS3 -> ds_load path).
__global__ __launch_bounds__(256) void gemm_bf16(
    const unsigned short* __restrict__ X,
    const unsigned short* __restrict__ Wt,
    float* __restrict__ C, int M, int N, int K)
{
  __shared__ unsigned short Xs[2][256 * 64];   // 2 x 32 KB
  __shared__ unsigned short Ws[2][128 * 64];   // 2 x 16 KB  (96 KB total; WGP LDS = 320 KB)

  const int tid  = threadIdx.x;
  const int m0   = blockIdx.y * 256, n0 = blockIdx.x * 128;
  const int wave = tid >> 5, lane = tid & 31;
  const int wm   = wave >> 1, wn = wave & 1;       // 4 x 2 wave grid
  const int half = lane >> 4, l15 = lane & 15;
  const v8f zero = {0.f,0.f,0.f,0.f,0.f,0.f,0.f,0.f};
  v8f acc[4][4];
  #pragma unroll
  for (int i = 0; i < 4; ++i)
    #pragma unroll
    for (int j = 0; j < 4; ++j) acc[i][j] = zero;

  auto stage = [&](int k0, int bufi) {
    // X tile: 256 rows x 128 B  (2048 x 16B chunks, 8 per thread)
    unsigned xbase = (unsigned)(size_t)&Xs[bufi][0];
    #pragma unroll
    for (int i = 0; i < 8; ++i) {
      int c = tid + 256 * i;               // 0..2047
      int row = c >> 3, seg = c & 7;
      async_b128(xbase + row * 128 + seg * 16,
                 (const char*)X + ((size_t)(m0 + row) * K + k0) * 2 + seg * 16);
    }
    // W tile: 128 rows (n) x 128 B (k)  (1024 x 16B chunks, 4 per thread)
    unsigned wbase = (unsigned)(size_t)&Ws[bufi][0];
    #pragma unroll
    for (int i = 0; i < 4; ++i) {
      int c = tid + 256 * i;               // 0..1023
      int row = c >> 3, seg = c & 7;
      async_b128(wbase + row * 128 + seg * 16,
                 (const char*)Wt + ((size_t)(n0 + row) * K + k0) * 2 + seg * 16);
    }
  };

  int buf = 0;
  stage(0, 0);
  for (int k0 = 0; k0 < K; k0 += 64) {
    wait_async0();
    __syncthreads();
    if (k0 + 64 < K) stage(k0 + 64, buf ^ 1);

    const unsigned* Xu = (const unsigned*)&Xs[buf][0];   // row stride 32 uints
    const unsigned* Wu = (const unsigned*)&Ws[buf][0];   // row stride 32 uints
    #pragma unroll
    for (int kc = 0; kc < 64; kc += 32) {
      Frag a[4], b[4];
      #pragma unroll
      for (int t = 0; t < 4; ++t) {
        int row = wm * 64 + t * 16 + l15;
        #pragma unroll
        for (int vv = 0; vv < 8; ++vv)     // A: K = 16*(v>>2)+8*half+2*(v&3)+j
          a[t].u[vv] = Xu[row * 32 + (kc >> 1) + 8 * (vv >> 2) + 4 * half + (vv & 3)];
      }
      #pragma unroll
      for (int t = 0; t < 4; ++t) {
        int col = wn * 64 + t * 16 + l15;
        #pragma unroll
        for (int vv = 0; vv < 8; ++vv)     // B: K = 16*(v>>2)+8*half+2*(v&3)+j
          b[t].u[vv] = Wu[col * 32 + (kc >> 1) + 8 * (vv >> 2) + 4 * half + (vv & 3)];
      }
      #pragma unroll
      for (int tm = 0; tm < 4; ++tm)
        #pragma unroll
        for (int tn = 0; tn < 4; ++tn)
          acc[tm][tn] = __builtin_amdgcn_wmma_f32_16x16x32_bf16(
              false, a[tm].v, false, b[tn].v, (short)0, acc[tm][tn], false, false);
    }
    buf ^= 1;
  }
  #pragma unroll
  for (int tm = 0; tm < 4; ++tm)
    for (int tn = 0; tn < 4; ++tn)
      for (int vv = 0; vv < 8; ++vv) {
        int row = m0 + wm * 64 + tm * 16 + vv + 8 * half;
        int col = n0 + wn * 64 + tn * 16 + l15;
        C[(size_t)row * N + col] = acc[tm][tn][vv];
      }
}

// ---------------------------------------------------------------- RoPE + rank contraction
// A-projections are padded to stride 128. v is written transposed [B,H,D,S].
__global__ __launch_bounds__(128) void build_qkv(
    const float* __restrict__ Aq, const float* __restrict__ Ak, const float* __restrict__ Av,
    const float* __restrict__ Bq, const float* __restrict__ Bk, const float* __restrict__ Bv,
    const float* __restrict__ fc, const float* __restrict__ fs,
    unsigned short* __restrict__ q, unsigned short* __restrict__ k,
    unsigned short* __restrict__ vT)
{
  const int bs = blockIdx.x;                  // b*2048 + s
  const int b = bs >> 11, s = bs & 2047;
  const int p = threadIdx.x;                  // pair index 0..127
  const float c = fc[s * 128 + p], sn = fs[s * 128 + p];
  float qe[6], qo[6], ke[2], ko[2], ve[2], vo[2];
  #pragma unroll
  for (int r = 0; r < 6; ++r) {
    float e = Bq[((size_t)bs * 6 + r) * 256 + 2 * p];
    float o = Bq[((size_t)bs * 6 + r) * 256 + 2 * p + 1];
    qe[r] = e * c - o * sn; qo[r] = e * sn + o * c;
  }
  #pragma unroll
  for (int r = 0; r < 2; ++r) {
    float e = Bk[((size_t)bs * 2 + r) * 256 + 2 * p];
    float o = Bk[((size_t)bs * 2 + r) * 256 + 2 * p + 1];
    ke[r] = e * c - o * sn; ko[r] = e * sn + o * c;
    ve[r] = Bv[((size_t)bs * 2 + r) * 256 + 2 * p];
    vo[r] = Bv[((size_t)bs * 2 + r) * 256 + 2 * p + 1];
  }
  #pragma unroll
  for (int h = 0; h < 8; ++h) {
    float aqe = 0.f, aqo = 0.f, ake = 0.f, ako = 0.f, ave = 0.f, avo = 0.f;
    #pragma unroll
    for (int r = 0; r < 6; ++r) {
      float a = Aq[(size_t)bs * 128 + h * 6 + r];
      aqe += a * qe[r]; aqo += a * qo[r];
    }
    #pragma unroll
    for (int r = 0; r < 2; ++r) {
      float a = Ak[(size_t)bs * 128 + h * 2 + r];
      ake += a * ke[r]; ako += a * ko[r];
      float a2 = Av[(size_t)bs * 128 + h * 2 + r];
      ave += a2 * ve[r]; avo += a2 * vo[r];
    }
    size_t o = (((size_t)(b * 8 + h)) * 2048 + s) * 256 + 2 * p;
    q[o]     = f2bf(aqe * (SCALING_F / 6.0f));   // fold scaling + 1/Q_RANK
    q[o + 1] = f2bf(aqo * (SCALING_F / 6.0f));
    k[o]     = f2bf(ake * 0.5f);                 // 1/K_RANK
    k[o + 1] = f2bf(ako * 0.5f);
    size_t ot = (((size_t)(b * 8 + h)) * 256 + 2 * p) * 2048 + s;   // [B,H,D,S]
    vT[ot]        = f2bf(ave * 0.5f);            // 1/V_RANK
    vT[ot + 2048] = f2bf(avo * 0.5f);
  }
}

// ---------------------------------------------------------------- attention (softcap + causal)
// One workgroup (8 waves) per (b, h, 16-row q tile). 221 KB LDS score buffer (WGP LDS=320KB).
__global__ __launch_bounds__(256) void attn_kernel(
    const unsigned short* __restrict__ qg,
    const unsigned short* __restrict__ kg,
    const unsigned short* __restrict__ vtg,    // [B,H,D,S]
    unsigned short* __restrict__ ao)
{
  extern __shared__ char smem[];
  unsigned*       qs   = (unsigned*)smem;                               // 16x256 bf16  (8 KB)
  float*          sc   = (float*)(smem + 8192);                          // 16x2048 f32 (128 KB)
  unsigned short* pb   = (unsigned short*)(smem + 8192 + 131072);        // 16x2048 bf16 (64 KB)
  unsigned short* vt   = pb + 16 * 2048;                                 // 256x32 bf16  (16 KB)
  float*          rsum = (float*)(vt + 256 * 32);                        // 16 f32

  const int tid = threadIdx.x;
  const int wave = tid >> 5, lane = tid & 31;
  const int half = lane >> 4, l15 = lane & 15;
  const int qt = blockIdx.x, bh = blockIdx.y;
  const int b = bh >> 3, h = bh & 7;
  const int q0 = qt * 16;
  const size_t base = (size_t)bh * 2048 * 256;     // element base in [B,H,S,D]
  const unsigned* ku = (const unsigned*)kg;
  const v8f zero = {0.f,0.f,0.f,0.f,0.f,0.f,0.f,0.f};

  // stage q tile (16 rows x 512 B) via async copies: 512 chunks, 2 per thread
  {
    unsigned qbase = (unsigned)(size_t)qs;
    #pragma unroll
    for (int i = 0; i < 2; ++i) {
      int cix = tid + 256 * i;                 // 0..511
      int row = cix >> 5, seg = cix & 31;
      async_b128(qbase + row * 512 + seg * 16,
                 (const char*)qg + (base + (size_t)(q0 + row) * 256) * 2 + seg * 16);
    }
  }
  const int klim = q0 + 16;                      // causal bound (multiple of 16)
  const int kpad = (klim + 31) & ~31;            // multiple of 32 for PV chunks
  wait_async0();
  __syncthreads();

  // ---- pass A: scores = softcap(q.kT) + mask -> sc
  // hoist all 8 q A-fragments (k-tile invariant)
  Frag aq[8];
  #pragma unroll
  for (int d8 = 0; d8 < 8; ++d8)
    #pragma unroll
    for (int vv = 0; vv < 8; ++vv)
      aq[d8].u[vv] = qs[l15 * 128 + d8 * 16 + 8 * (vv >> 2) + 4 * half + (vv & 3)];

  for (int kt = wave; kt * 16 < klim; kt += 8) {
    v8f accv = zero;
    int krow = kt * 16 + l15;                    // B column n = k row (kT)
    #pragma unroll
    for (int d8 = 0; d8 < 8; ++d8) {
      Frag bb;
      #pragma unroll
      for (int vv = 0; vv < 8; ++vv)
        bb.u[vv] = ku[(base >> 1) + (size_t)krow * 128 + d8 * 16 + 8 * (vv >> 2) + 4 * half + (vv & 3)];
      accv = __builtin_amdgcn_wmma_f32_16x16x32_bf16(
          false, aq[d8].v, false, bb.v, (short)0, accv, false, false);
    }
    #pragma unroll
    for (int vv = 0; vv < 8; ++vv) {
      int qi = q0 + vv + 8 * half;
      int ki = kt * 16 + l15;
      float sval = tanhf(accv[vv] * (1.0f / SOFTCAP_F)) * SOFTCAP_F;
      if (ki > qi) sval = NEGV;
      sc[(vv + 8 * half) * 2048 + ki] = sval;
    }
  }
  __syncthreads();

  // ---- pass B: row softmax (16 lanes per row), unnormalized probs in bf16
  {
    int row = tid >> 4, ls = tid & 15;
    float m = -3.0e38f;
    for (int cc = ls; cc < klim; cc += 16) m = fmaxf(m, sc[row * 2048 + cc]);
    #pragma unroll
    for (int off = 1; off < 16; off <<= 1) m = fmaxf(m, __shfl_xor(m, off, 32));
    float ssum = 0.f;
    for (int cc = ls; cc < klim; cc += 16) {
      float e = __expf(sc[row * 2048 + cc] - m);
      pb[row * 2048 + cc] = f2bf(e);
      ssum += e;
    }
    #pragma unroll
    for (int off = 1; off < 16; off <<= 1) ssum += __shfl_xor(ssum, off, 32);
    if (ls == 0) rsum[row] = ssum;
    for (int cc = klim + ls; cc < kpad; cc += 16) pb[row * 2048 + cc] = 0;  // pad chunk
  }
  __syncthreads();

  // ---- pass C: out = P.V ; wave w owns d-range [w*32, w*32+32)
  v8f oacc[2]; oacc[0] = zero; oacc[1] = zero;
  const unsigned* pbu = (const unsigned*)pb;
  const unsigned* vtu = (const unsigned*)vt;
  const unsigned vtbase = (unsigned)(size_t)vt;
  for (int kc = 0; kc < kpad; kc += 32) {
    __syncthreads();                           // previous chunk fully consumed
    #pragma unroll
    for (int i = 0; i < 4; ++i) {              // stage vT chunk: 256 rows x 64 B, async
      int cix = tid + 256 * i;                 // 0..1023
      int d = cix >> 2, seg = cix & 3;
      async_b128(vtbase + d * 64 + seg * 16,
                 (const char*)vtg + (((size_t)bh * 256 + d) * 2048 + kc) * 2 + seg * 16);
    }
    wait_async0();
    __syncthreads();
    Frag a;
    #pragma unroll
    for (int vv = 0; vv < 8; ++vv)
      a.u[vv] = pbu[l15 * 1024 + (kc >> 1) + 8 * (vv >> 2) + 4 * half + (vv & 3)];
    #pragma unroll
    for (int tn = 0; tn < 2; ++tn) {
      Frag bb;
      int d = wave * 32 + tn * 16 + l15;
      #pragma unroll
      for (int vv = 0; vv < 8; ++vv)
        bb.u[vv] = vtu[d * 16 + 8 * (vv >> 2) + 4 * half + (vv & 3)];
      oacc[tn] = __builtin_amdgcn_wmma_f32_16x16x32_bf16(
          false, a.v, false, bb.v, (short)0, oacc[tn], false, false);
    }
  }
  #pragma unroll
  for (int tn = 0; tn < 2; ++tn)
    for (int vv = 0; vv < 8; ++vv) {
      int row = vv + 8 * half;
      int d = wave * 32 + tn * 16 + l15;
      float val = oacc[tn][vv] / rsum[row];
      ao[((size_t)(b * 2048 + q0 + row)) * 2048 + h * 256 + d] = f2bf(val);
    }
}

// ---------------------------------------------------------------- host
extern "C" void kernel_launch(void* const* d_in, const int* in_sizes, int n_in,
                              void* d_out, int out_size, void* d_ws, size_t ws_size,
                              hipStream_t stream) {
  (void)in_sizes; (void)n_in; (void)out_size; (void)ws_size;
  const float* hs  = (const float*)d_in[0];
  const float* fc  = (const float*)d_in[1];
  const float* fs  = (const float*)d_in[2];
  // d_in[3] = mask (unused; causal mask applied analytically)
  const float* WAq = (const float*)d_in[4];
  const float* WAk = (const float*)d_in[5];
  const float* WAv = (const float*)d_in[6];
  const float* WBq = (const float*)d_in[7];
  const float* WBk = (const float*)d_in[8];
  const float* WBv = (const float*)d_in[9];
  const float* Wo  = (const float*)d_in[10];
  float* out = (float*)d_out;

  char* ws = (char*)d_ws;
  size_t off = 0;
  auto alloc = [&](size_t bytes) { size_t o = off; off += (bytes + 255) & ~(size_t)255; return o; };

  const int BS = 4096, K = 2048;
  unsigned short* xb  = (unsigned short*)(ws + alloc((size_t)BS * K * 2));
  // weights: bf16, TRANSPOSED [Npad, K]
  unsigned short* wAq = (unsigned short*)(ws + alloc((size_t)128  * K * 2));
  unsigned short* wAk = (unsigned short*)(ws + alloc((size_t)128  * K * 2));
  unsigned short* wAv = (unsigned short*)(ws + alloc((size_t)128  * K * 2));
  unsigned short* wBq = (unsigned short*)(ws + alloc((size_t)1536 * K * 2));
  unsigned short* wBk = (unsigned short*)(ws + alloc((size_t)512  * K * 2));
  unsigned short* wBv = (unsigned short*)(ws + alloc((size_t)512  * K * 2));
  unsigned short* wO  = (unsigned short*)(ws + alloc((size_t)2048 * K * 2));
  // projection outputs, f32 (A-projections padded to stride 128)
  float* pAq = (float*)(ws + alloc((size_t)BS * 128 * 4));
  float* pAk = (float*)(ws + alloc((size_t)BS * 128 * 4));
  float* pAv = (float*)(ws + alloc((size_t)BS * 128 * 4));
  float* pBq = (float*)(ws + alloc((size_t)BS * 1536 * 4));
  float* pBk = (float*)(ws + alloc((size_t)BS * 512 * 4));
  float* pBv = (float*)(ws + alloc((size_t)BS * 512 * 4));
  unsigned short* qb  = (unsigned short*)(ws + alloc((size_t)BS * 2048 * 2));
  unsigned short* kb  = (unsigned short*)(ws + alloc((size_t)BS * 2048 * 2));
  unsigned short* vtb = (unsigned short*)(ws + alloc((size_t)BS * 2048 * 2));  // [B,H,D,S]
  unsigned short* aob = (unsigned short*)(ws + alloc((size_t)BS * 2048 * 2));

  {
    int n = BS * K;
    cvt_f32_bf16<<<(n + 2047) / 2048, 256, 0, stream>>>(hs, xb, n);
  }
  auto cvtT = [&](const float* s, unsigned short* d, int Nsrc, int Npad) {
    int total = K * Npad;
    cvt_transpose_pad<<<(total + 2047) / 2048, 256, 0, stream>>>(s, d, K, Nsrc, Npad);
  };
  cvtT(WAq, wAq, 48, 128);    cvtT(WAk, wAk, 16, 128);   cvtT(WAv, wAv, 16, 128);
  cvtT(WBq, wBq, 1536, 1536); cvtT(WBk, wBk, 512, 512);  cvtT(WBv, wBv, 512, 512);
  cvtT(Wo,  wO,  2048, 2048);

  auto gemm = [&](const unsigned short* X, const unsigned short* Wt, float* C, int N) {
    dim3 grid(N / 128, BS / 256);
    gemm_bf16<<<grid, 256, 0, stream>>>(X, Wt, C, BS, N, K);
  };
  gemm(xb, wAq, pAq, 128);
  gemm(xb, wAk, pAk, 128);
  gemm(xb, wAv, pAv, 128);
  gemm(xb, wBq, pBq, 1536);
  gemm(xb, wBk, pBk, 512);
  gemm(xb, wBv, pBv, 512);

  build_qkv<<<BS, 128, 0, stream>>>(pAq, pAk, pAv, pBq, pBk, pBv, fc, fs, qb, kb, vtb);

  const size_t attn_smem = 8192 + 131072 + 65536 + 16384 + 64;   // 221,248 B
  (void)hipFuncSetAttribute((const void*)attn_kernel,
                            hipFuncAttributeMaxDynamicSharedMemorySize, (int)attn_smem);
  attn_kernel<<<dim3(128, 16), 256, attn_smem, stream>>>(qb, kb, vtb, aob);

  // final: out = attn_out @ W_o  (4096x2048 x 2048x2048, f32 result)
  {
    dim3 grid(2048 / 128, BS / 256);
    gemm_bf16<<<grid, 256, 0, stream>>>(aob, wO, out, BS, 2048, K);
  }
}
